// FuzzyMealyMachine_13022340842136
// MI455X (gfx1250) — compile-verified, matrix operand order checked by reading, not run on hardware
//
#include <hip/hip_runtime.h>
#include <hip/hip_bf16.h>

#define S       512
#define NI      256
#define L       4096
#define CHUNK   64
#define NC      (L / CHUNK)     // 64 chunks
#define RB      8               // row-blocks per chunk
#define BROWS   (S / RB)        // 64 rows of M owned per block

// LDS byte offsets inside K2's (single) dynamic LDS allocation
#define MBUF_BYTES   (BROWS * S * 2)            // 64 KB per M buffer
#define BSTRIP_BYTES (32 * S * 2)               // 32 KB per B strip
#define BOFF0        (2 * MBUF_BYTES)           // 131072
#define K2_LDS       (2 * MBUF_BYTES + 2 * BSTRIP_BYTES)   // 192 KB

typedef __attribute__((ext_vector_type(16))) __bf16  v16bf;
typedef __attribute__((ext_vector_type(8)))  float   v8f;

union BfTile { uint4 q[2]; v16bf v; };

__device__ __forceinline__ unsigned short f2bf(float f) {
    unsigned int u = __float_as_uint(f);
    u += 0x7FFFu + ((u >> 16) & 1u);          // round-to-nearest-even
    return (unsigned short)(u >> 16);
}
__device__ __forceinline__ float bf2f(unsigned int h) {
    return __uint_as_float(h << 16);
}

#ifndef __has_builtin
#define __has_builtin(x) 0
#endif
#if __has_builtin(__builtin_amdgcn_tensor_load_to_lds) && \
    __has_builtin(__builtin_amdgcn_s_wait_tensorcnt)
#define USE_TDM 1
#else
#define USE_TDM 0
#endif

#if USE_TDM
typedef unsigned int u32x4 __attribute__((ext_vector_type(4)));
typedef int          i32x4 __attribute__((ext_vector_type(4)));
typedef int          i32x8 __attribute__((ext_vector_type(8)));

// TDM 2D tile load: 512 rows x 32 bf16 (64B) per row, row stride 1KB, into a
// contiguous [n][k] 32KB LDS strip.  One descriptor == whole strip; tracked
// with TENSORcnt so it overlaps WMMA on the other buffer.
__device__ __forceinline__ void tdm_stage_strip(unsigned lds_byte_addr,
                                                const unsigned short* gsrc) {
    unsigned long long ga = (unsigned long long)(uintptr_t)gsrc;
    unsigned ga_lo = __builtin_amdgcn_readfirstlane((unsigned)(ga & 0xFFFFFFFFu));
    unsigned ga_hi = __builtin_amdgcn_readfirstlane((unsigned)(ga >> 32));
    u32x4 g0;
    g0[0] = 1u;                                    // count=1, user descriptor
    g0[1] = lds_byte_addr;                         // D#.lds_addr
    g0[2] = ga_lo;                                 // D#.global_addr[31:0]
    g0[3] = (ga_hi & 0x01FFFFFFu) | (2u << 30);    // global_addr[56:32] | type=2
    i32x8 g1;
    g1[0] = 0x10000;          // workgroup_mask=0, data_size=1 (2B elements)
    g1[1] = (32 << 16);       // tensor_dim0 = 32
    g1[2] = (int)(512u << 16);// tensor_dim1 = 512
    g1[3] = (32 << 16);       // tile_dim0 = 32
    g1[4] = 512;              // tile_dim1 = 512, tile_dim2 = 0
    g1[5] = 512;              // tensor_dim0_stride = 512 elements (1KB)
    g1[6] = 0;
    g1[7] = 0;
    i32x4 z4 = {0, 0, 0, 0};
#if __has_include(<hip/amd_detail/amd_gfx1250_TDM.h>)
    i32x8 z8 = {0, 0, 0, 0, 0, 0, 0, 0};
    __builtin_amdgcn_tensor_load_to_lds(g0, g1, z4, z4, z8, 0);
#else
    __builtin_amdgcn_tensor_load_to_lds(g0, g1, z4, z4, 0);
#endif
}
#endif // USE_TDM

// ---------------------------------------------------------------------------
// K1: softmax over last axis, emit bf16 TRANSPOSED per input: out[i][col][s].
// ---------------------------------------------------------------------------
__global__ void softmax_transpose_kernel(const float* __restrict__ Tlog,
                                         const float* __restrict__ Olog,
                                         unsigned short* __restrict__ Tb,
                                         unsigned short* __restrict__ Ob) {
    __shared__ unsigned short tile_t[64 * S];  // [col][row] : 64 KB
    const int b      = blockIdx.x;             // 0..4095
    const int tensor = b >> 11;                // 0=T, 1=O
    const int rem    = b & 2047;
    const int i      = rem >> 3;
    const int s0     = (rem & 7) * 64;
    const float* src = tensor ? Olog : Tlog;
    unsigned short* dst = tensor ? Ob : Tb;

    const int tid = threadIdx.x;
    const int r   = tid >> 2;                  // local row 0..63
    const int qq  = tid & 3;                   // column quarter
    const int s   = s0 + r;
    const float* row = src + (size_t)(s * NI + i) * S + qq * 128;

    float m = -3.4e38f;
    for (int j = 0; j < 32; ++j) {
        float4 v = ((const float4*)row)[j];
        m = fmaxf(m, fmaxf(fmaxf(v.x, v.y), fmaxf(v.z, v.w)));
    }
    m = fmaxf(m, __shfl_xor(m, 1));
    m = fmaxf(m, __shfl_xor(m, 2));

    float sum = 0.f;
    for (int j = 0; j < 32; ++j) {
        float4 v = ((const float4*)row)[j];
        sum += __expf(v.x - m) + __expf(v.y - m) + __expf(v.z - m) + __expf(v.w - m);
    }
    sum += __shfl_xor(sum, 1);
    sum += __shfl_xor(sum, 2);
    const float inv = 1.0f / sum;

    for (int j = 0; j < 32; ++j) {
        float4 v = ((const float4*)row)[j];
        int c = qq * 128 + j * 4;
        tile_t[(c + 0) * 64 + r] = f2bf(__expf(v.x - m) * inv);
        tile_t[(c + 1) * 64 + r] = f2bf(__expf(v.y - m) * inv);
        tile_t[(c + 2) * 64 + r] = f2bf(__expf(v.z - m) * inv);
        tile_t[(c + 3) * 64 + r] = f2bf(__expf(v.w - m) * inv);
    }
    __syncthreads();

    // coalesced transposed store: dst[(i*S + c)*S + s0 + k], 128B per column
    const uint4* tq = (const uint4*)tile_t;
    #pragma unroll
    for (int u = 0; u < 16; ++u) {
        int mIdx = tid + u * 256;
        int c = mIdx >> 3, oct = mIdx & 7;
        *(((uint4*)(dst + (size_t)(i * S + c) * S + s0)) + oct) = tq[c * 8 + oct];
    }
}

// ---------------------------------------------------------------------------
// K2: chunk transition matrices via WMMA + TDM-pipelined B staging.
// Block owns 64 rows of M (identity init), iterates M <- M @ T_t, 64 steps.
// LDS: M double buffer (2x64KB) + double-buffered 32KB T_t K-strips.
// ---------------------------------------------------------------------------
__global__ void chunk_matmul_kernel(const unsigned short* __restrict__ Tb,
                                    const int* __restrict__ seq,
                                    unsigned short* __restrict__ Mc) {
    extern __shared__ unsigned char smem[];
    unsigned short* Mbuf0 = (unsigned short*)smem;
    unsigned short* Mbuf1 = Mbuf0 + BROWS * S;
    unsigned short* Bst0  = Mbuf1 + BROWS * S;
    unsigned short* Bst1  = Bst0 + 32 * S;

    const int tid   = threadIdx.x;
    const int lane  = tid & 31;
    const int w     = tid >> 5;            // wave 0..7
    const int rt    = w >> 1;              // local row-tile 0..3
    const int ch    = w & 1;               // column half 0..1
    const int chunk = blockIdx.x / RB;
    const int rbase = (blockIdx.x % RB) * BROWS;
    const int lhalf = lane >> 4;           // lane-half K split (ISA 7.12.2)
    const int lmod  = lane & 15;
    const int k8    = lhalf << 3;

    unsigned short* Mold = Mbuf0;
    unsigned short* Mnew = Mbuf1;

    for (int m = tid; m < BROWS * S; m += 256) {
        int r = m >> 9, k = m & (S - 1);
        Mold[m] = ((rbase + r) == k) ? (unsigned short)0x3F80 : (unsigned short)0;
    }
    __syncthreads();

    for (int step = 0; step < CHUNK; ++step) {
        const unsigned short* Tt = Tb + (size_t)seq[chunk * CHUNK + step] * S * S;
        if (step + 1 < CHUNK)
            __builtin_prefetch(Tb + (size_t)seq[chunk * CHUNK + step + 1] * S * S, 0, 1);

        for (int cg = 0; cg < 2; ++cg) {
            v8f acc[8] = {};

            // stage strip kb=0 into buffer 0
#if USE_TDM
            if (w == 0) tdm_stage_strip(BOFF0, Tt);
#else
            {
                uint4 tmp[8];
                #pragma unroll
                for (int o = 0; o < 8; ++o) {
                    int mm = tid + o * 256, n = mm >> 2, oo = mm & 3;
                    tmp[o] = *(const uint4*)(Tt + (size_t)n * S + oo * 8);
                }
                uint4* bq = (uint4*)Bst0;
                #pragma unroll
                for (int o = 0; o < 8; ++o) bq[tid + o * 256] = tmp[o];
            }
#endif
            for (int kb = 0; kb < 16; ++kb) {
                __syncthreads();   // previous compute done; other buffer free
#if USE_TDM
                if (w == 0) {
                    if (kb < 15) {
                        tdm_stage_strip(BOFF0 + (((kb + 1) & 1) ? BSTRIP_BYTES : 0),
                                        Tt + (kb + 1) * 32);
                        __builtin_amdgcn_s_wait_tensorcnt(1);  // strip kb done
                    } else {
                        __builtin_amdgcn_s_wait_tensorcnt(0);
                    }
                }
#else
                if (kb < 15) {
                    uint4 tmp[8];
                    #pragma unroll
                    for (int o = 0; o < 8; ++o) {
                        int mm = tid + o * 256, n = mm >> 2, oo = mm & 3;
                        tmp[o] = *(const uint4*)(Tt + (size_t)n * S + (kb + 1) * 32 + oo * 8);
                    }
                    uint4* bq = (uint4*)(((kb + 1) & 1) ? Bst1 : Bst0);
                    #pragma unroll
                    for (int o = 0; o < 8; ++o) bq[tid + o * 256] = tmp[o];
                }
#endif
                __syncthreads();   // strip kb visible

                // A = M_old tile (16 rows x 32 K) in ISA lane layout
                BfTile A;
                const unsigned short* ap = Mold + (rt * 16 + lmod) * S + kb * 32 + k8;
                A.q[0] = *(const uint4*)(ap);
                A.q[1] = *(const uint4*)(ap + 16);

                // batch-load all 8 B tiles, then stream 8 WMMAs back-to-back
                const unsigned short* Bb = (kb & 1) ? Bst1 : Bst0;
                BfTile Bt[8];
                #pragma unroll
                for (int j = 0; j < 8; ++j) {
                    const int n = (ch * 16 + cg * 8 + j) * 16 + lmod;
                    const unsigned short* bp = Bb + n * 32 + k8;
                    Bt[j].q[0] = *(const uint4*)(bp);
                    Bt[j].q[1] = *(const uint4*)(bp + 16);
                }
                #pragma unroll
                for (int j = 0; j < 8; ++j)
                    acc[j] = __builtin_amdgcn_wmma_f32_16x16x32_bf16(
                        false, A.v, false, Bt[j].v, (short)0, acc[j], false, false);
            }

            // D layout: vgpr v -> row v + 8*(lane>=16); col = lane&15
            #pragma unroll
            for (int j = 0; j < 8; ++j) {
                const int n = (ch * 16 + cg * 8 + j) * 16 + lmod;
                #pragma unroll
                for (int vg = 0; vg < 8; ++vg) {
                    const int mrow = rt * 16 + (lhalf << 3) + vg;
                    Mnew[mrow * S + n] = f2bf(acc[j][vg]);
                }
            }
        }
        __syncthreads();
        unsigned short* tmp = Mold; Mold = Mnew; Mnew = tmp;
    }

    // write this block's 64 rows of the chunk matrix (bf16, row-major [s][s'])
    uint4* dq = (uint4*)(Mc + ((size_t)chunk * S + rbase) * S);
    const uint4* sq = (const uint4*)Mold;
    #pragma unroll
    for (int u = 0; u < 16; ++u) dq[tid + u * 256] = sq[tid + u * 256];
}

// ---------------------------------------------------------------------------
// K3: sequential chunk-boundary propagation: state <- state @ M_c (64 steps).
// ---------------------------------------------------------------------------
__global__ void boundary_kernel(const unsigned short* __restrict__ Mc,
                                const int* __restrict__ initial_state,
                                float* __restrict__ bstate) {
    __shared__ float st[S];
    const int tid = threadIdx.x;           // 512 threads
    const int is0 = *initial_state;
    st[tid] = (tid == is0) ? 1.f : 0.f;
    bstate[tid] = st[tid];
    __syncthreads();
    for (int c = 0; c < NC; ++c) {
        const unsigned short* M = Mc + (size_t)c * S * S;
        float acc = 0.f;
        #pragma unroll 8
        for (int s = 0; s < S; ++s)
            acc = fmaf(st[s], bf2f(M[(size_t)s * S + tid]), acc);
        __syncthreads();
        st[tid] = acc;
        bstate[(size_t)(c + 1) * S + tid] = acc;
        __syncthreads();
    }
}

// ---------------------------------------------------------------------------
// K4: per-chunk replay, fused output + state matvecs (HBM-bandwidth bound).
// ---------------------------------------------------------------------------
__global__ void replay_kernel(const unsigned short* __restrict__ Tb,
                              const unsigned short* __restrict__ Ob,
                              const int* __restrict__ seq,
                              const float* __restrict__ bstate,
                              float* __restrict__ out) {
    __shared__ float stA[S], stB[S];
    const int tid   = threadIdx.x;         // 256 threads
    const int chunk = blockIdx.x;
    stA[tid]       = bstate[(size_t)chunk * S + tid];
    stA[tid + 256] = bstate[(size_t)chunk * S + tid + 256];

    for (int step = 0; step < CHUNK; ++step) {
        __syncthreads();
        const int t  = chunk * CHUNK + step;
        const int it = seq[t];
        const unsigned short* Orow = Ob + (size_t)it * S * S;
        const unsigned short* Trow = Tb + (size_t)it * S * S;
        if (step + 1 < CHUNK)
            __builtin_prefetch(Tb + (size_t)seq[t + 1] * S * S, 0, 1);
        float* cur = (step & 1) ? stB : stA;
        float* nxt = (step & 1) ? stA : stB;

        #pragma unroll
        for (int h = 0; h < 2; ++h) {
            const int col = tid + h * 256;
            float aO = 0.f, aT = 0.f;
            const uint4* po = (const uint4*)(Orow + (size_t)col * S);
            const uint4* pt = (const uint4*)(Trow + (size_t)col * S);
            #pragma unroll 4
            for (int j = 0; j < 64; ++j) {
                uint4 qo = po[j];
                uint4 qt = pt[j];
                const float* sp = &cur[j * 8];
                aO = fmaf(sp[0], bf2f(qo.x & 0xFFFFu), aO);
                aO = fmaf(sp[1], bf2f(qo.x >> 16),    aO);
                aO = fmaf(sp[2], bf2f(qo.y & 0xFFFFu), aO);
                aO = fmaf(sp[3], bf2f(qo.y >> 16),    aO);
                aO = fmaf(sp[4], bf2f(qo.z & 0xFFFFu), aO);
                aO = fmaf(sp[5], bf2f(qo.z >> 16),    aO);
                aO = fmaf(sp[6], bf2f(qo.w & 0xFFFFu), aO);
                aO = fmaf(sp[7], bf2f(qo.w >> 16),    aO);
                aT = fmaf(sp[0], bf2f(qt.x & 0xFFFFu), aT);
                aT = fmaf(sp[1], bf2f(qt.x >> 16),    aT);
                aT = fmaf(sp[2], bf2f(qt.y & 0xFFFFu), aT);
                aT = fmaf(sp[3], bf2f(qt.y >> 16),    aT);
                aT = fmaf(sp[4], bf2f(qt.z & 0xFFFFu), aT);
                aT = fmaf(sp[5], bf2f(qt.z >> 16),    aT);
                aT = fmaf(sp[6], bf2f(qt.w & 0xFFFFu), aT);
                aT = fmaf(sp[7], bf2f(qt.w >> 16),    aT);
            }
            out[(size_t)t * S + col] = aO;
            nxt[col] = aT;
        }
    }
}

extern "C" void kernel_launch(void* const* d_in, const int* in_sizes, int n_in,
                              void* d_out, int out_size, void* d_ws, size_t ws_size,
                              hipStream_t stream) {
    (void)in_sizes; (void)n_in; (void)out_size; (void)ws_size;
    const float* Tlog = (const float*)d_in[0];
    const float* Olog = (const float*)d_in[1];
    const int*   seq  = (const int*)d_in[2];
    const int*   ist  = (const int*)d_in[3];
    float* out = (float*)d_out;

    // workspace layout: Tb(128MB) | Ob(128MB) | Mc(32MB) | bstate(130KB)
    unsigned short* Tb = (unsigned short*)d_ws;
    unsigned short* Ob = Tb + (size_t)S * NI * S;
    unsigned short* Mc = Ob + (size_t)S * NI * S;
    float* bstate = (float*)(Mc + (size_t)NC * S * S);

    softmax_transpose_kernel<<<2 * NI * (S / 64), 256, 0, stream>>>(Tlog, Olog, Tb, Ob);

    hipFuncSetAttribute((const void*)chunk_matmul_kernel,
                        hipFuncAttributeMaxDynamicSharedMemorySize, K2_LDS);
    chunk_matmul_kernel<<<NC * RB, 256, K2_LDS, stream>>>(Tb, seq, Mc);

    boundary_kernel<<<1, S, 0, stream>>>(Mc, ist, bstate);
    replay_kernel<<<NC, 256, 0, stream>>>(Tb, Ob, seq, bstate, out);
}
